// VectorQuantizer_78030965834031
// MI455X (gfx1250) — compile-verified, matrix operand order checked by reading, not run on hardware
//
#include <hip/hip_runtime.h>
#include <hip/hip_bf16.h>

#define KCODES 1024
#define DDIM   256
#define BBATCH 16
#define TTIME  4096
#define NROWS  (BBATCH * TTIME)          // 65536
#define Q_ELEMS (BBATCH * DDIM * TTIME)  // 16777216
#define LOSS_OFF Q_ELEMS
#define IDX_OFF  (Q_ELEMS + 1)
#define PERP_OFF (Q_ELEMS + 1 + NROWS)

typedef __attribute__((ext_vector_type(16))) __bf16         v16bf;
typedef __attribute__((ext_vector_type(8)))  float          v8f;
typedef __attribute__((ext_vector_type(16))) unsigned short v16u16;
typedef __attribute__((ext_vector_type(8)))  unsigned short v8u16;

__device__ __forceinline__ unsigned short f2bf(float f) {
  unsigned int u = __builtin_bit_cast(unsigned int, f);
  u += 0x7FFFu + ((u >> 16) & 1u);              // round-to-nearest-even
  return (unsigned short)(u >> 16);
}

// ---------------------------------------------------------------------------
// Kernel 1: zero ws accumulators, compute exact f32 row norms, bf16 codebook.
// ws layout (floats): [0]=loss acc, [16..1039]=counts, [1040..2063]=norms,
// byte 8256..: bf16 codebook (row-major K x D, 512 KB, L2-resident).
// ---------------------------------------------------------------------------
__global__ void vq_init(const float* __restrict__ emb,
                        float* __restrict__ ws_f,
                        unsigned short* __restrict__ emb_bf) {
  int k = blockIdx.x * blockDim.x + threadIdx.x;
  if (k == 0) ws_f[0] = 0.0f;
  if (k < KCODES) {
    ws_f[16 + k] = 0.0f;                        // counts
    const float* row = emb + (size_t)k * DDIM;
    unsigned short* orow = emb_bf + (size_t)k * DDIM;
    float s = 0.0f;
    for (int d = 0; d < DDIM; ++d) {
      float v = row[d];
      s += v * v;
      orow[d] = f2bf(v);
    }
    ws_f[1040 + k] = s;                         // ||e_k||^2 exact in f32
  }
}

// ---------------------------------------------------------------------------
// Kernel 2: bf16-WMMA distance GEMM + argmin + gather + loss + histogram.
// One wave owns 16 rows (16 consecutive t within one b). 8 waves / block.
// ---------------------------------------------------------------------------
__global__ __launch_bounds__(256)
void vq_main(const float* __restrict__ x,
             const float* __restrict__ emb,
             const float* __restrict__ norms,
             const unsigned short* __restrict__ emb_bf,
             float* __restrict__ out,
             float* __restrict__ counts,
             float* __restrict__ loss_acc) {
  __shared__ unsigned int s_idx[8 * 16];
  __shared__ float s_red[256];

  const int tid  = threadIdx.x;
  const int wid  = tid >> 5;
  const int lane = tid & 31;
  const int tl   = lane & 15;
  const bool hi  = (lane >= 16);

  const int n0 = (blockIdx.x * 8 + wid) * 16;   // first row of this wave's tile
  const int b  = n0 >> 12;                      // n0 / 4096
  const int t0 = n0 & 4095;

  const float* xp = x + (size_t)b * DDIM * TTIME + t0 + tl;

  // 16-bit A-fragment K mapping (ISA 7.12.2): lanes 0-15 hold K 0-7 & 16-23,
  // lanes 16-31 hold K 8-15 & 24-31 (per 32-wide k-step).
  const int klo = hi ? 8 : 0;
  const int khi = hi ? 24 : 16;

  // Load + convert the whole 16x256 A tile once; keep resident (64 VGPRs).
  v16bf afrag[8];
#pragma unroll
  for (int kt = 0; kt < 8; ++kt) {
    v16u16 au;
#pragma unroll
    for (int e = 0; e < 16; ++e) {
      int d = kt * 32 + ((e < 8) ? (klo + e) : (khi + e - 8));
      au[e] = f2bf(xp[(size_t)d * TTIME]);
    }
    afrag[kt] = __builtin_bit_cast(v16bf, au);
  }

  float bestV[8];
  unsigned int bestI[8];
#pragma unroll
  for (int j = 0; j < 8; ++j) { bestV[j] = 3.4e38f; bestI[j] = 0; }

  for (int ct = 0; ct < KCODES / 16; ++ct) {
    const int code = ct * 16 + tl;              // this lane's column/code
    const unsigned short* ep = emb_bf + (size_t)code * DDIM;
    if (ct + 1 < KCODES / 16)
      __builtin_prefetch(emb_bf + (size_t)(code + 16) * DDIM, 0, 1);

    v8f acc = {};
#pragma unroll
    for (int kt = 0; kt < 8; ++kt) {
      // B fragment: lane = code column; two contiguous 16B bf16 runs per k-step.
      v8u16 lo8 = *(const v8u16*)(ep + kt * 32 + klo);
      v8u16 hi8 = *(const v8u16*)(ep + kt * 32 + khi);
      v16u16 bu;
#pragma unroll
      for (int e = 0; e < 8; ++e) { bu[e] = lo8[e]; bu[e + 8] = hi8[e]; }
      v16bf bfrag = __builtin_bit_cast(v16bf, bu);
      acc = __builtin_amdgcn_wmma_f32_16x16x32_bf16(
          false, afrag[kt], false, bfrag, (short)0, acc, false, false);
    }

    const float nrm = norms[code];              // exact f32 correction term
#pragma unroll
    for (int j = 0; j < 8; ++j) {
      float s = nrm - 2.0f * acc[j];            // == dist - ||x||^2 (rank-equal)
      if (s < bestV[j]) { bestV[j] = s; bestI[j] = (unsigned)code; }
    }
  }

  // Min-reduce across the 16 lanes sharing each row (xor<16 stays in half-wave).
#pragma unroll
  for (int off = 1; off < 16; off <<= 1) {
#pragma unroll
    for (int j = 0; j < 8; ++j) {
      float ov        = __shfl_xor(bestV[j], off);
      unsigned int oi = (unsigned)__shfl_xor((int)bestI[j], off);
      if (ov < bestV[j] || (ov == bestV[j] && oi < bestI[j])) {
        bestV[j] = ov; bestI[j] = oi;
      }
    }
  }

  // C layout: VGPR j -> row j (lanes 0-15) / row j+8 (lanes 16-31).
  if (tl == 0) {
    const int rbase = hi ? 8 : 0;
#pragma unroll
    for (int j = 0; j < 8; ++j) {
      unsigned int id = bestI[j];
      s_idx[wid * 16 + rbase + j] = id;
      out[IDX_OFF + n0 + rbase + j] = (float)id; // indices emitted as f32
      atomicAdd(&counts[id], 1.0f);              // histogram for perplexity
    }
  }
  __syncthreads();

  // Gather quantized rows (exact f32), write q_out coalescedly, accumulate loss.
  const unsigned int myidx = s_idx[wid * 16 + tl];
  const float* erow = emb + (size_t)myidx * DDIM;
  float* qbase = out + (size_t)b * DDIM * TTIME + t0 + tl;
  const int dd = hi ? 1 : 0;                     // two d-planes per iteration
  float lsum = 0.0f;
#pragma unroll 4
  for (int d = dd; d < DDIM; d += 2) {
    float e  = erow[d];
    float xv = xp[(size_t)d * TTIME];
    qbase[(size_t)d * TTIME] = e;                // STE forward == quantized
    float df = e - xv;
    lsum += df * df;
  }

  s_red[tid] = lsum;
  __syncthreads();
  for (int s = 128; s > 0; s >>= 1) {
    if (tid < s) s_red[tid] += s_red[tid + s];
    __syncthreads();
  }
  if (tid == 0) atomicAdd(loss_acc, s_red[0]);
}

// ---------------------------------------------------------------------------
// Kernel 3: perplexity from counts; finalize loss = 1.25 * mean((q-x)^2).
// ---------------------------------------------------------------------------
__global__ void vq_final(const float* __restrict__ ws_f, float* __restrict__ out) {
  __shared__ float s_red[1024];
  int k = threadIdx.x;
  float p = ws_f[16 + k] * (1.0f / (float)NROWS);
  s_red[k] = p * logf(p + 1e-10f);
  __syncthreads();
  for (int s = 512; s > 0; s >>= 1) {
    if (k < s) s_red[k] += s_red[k + s];
    __syncthreads();
  }
  if (k == 0) {
    out[PERP_OFF] = expf(-s_red[0]);
    out[LOSS_OFF] = 1.25f * ws_f[0] / (float)Q_ELEMS;
  }
}

// ---------------------------------------------------------------------------
extern "C" void kernel_launch(void* const* d_in, const int* in_sizes, int n_in,
                              void* d_out, int out_size, void* d_ws, size_t ws_size,
                              hipStream_t stream) {
  const float* x   = (const float*)d_in[0];   // [16, 256, 4096] f32
  const float* emb = (const float*)d_in[1];   // [1024, 256] f32
  float* out  = (float*)d_out;
  float* ws_f = (float*)d_ws;
  float* counts = ws_f + 16;
  float* norms  = ws_f + 1040;
  unsigned short* emb_bf = (unsigned short*)((char*)d_ws + 8256); // 512 KB

  vq_init<<<4, 256, 0, stream>>>(emb, ws_f, emb_bf);
  vq_main<<<512, 256, 0, stream>>>(x, emb, norms, emb_bf, out, counts, ws_f);
  vq_final<<<1, 1024, 0, stream>>>(ws_f, out);
}
